// RNNWithMasking_73212012528181
// MI455X (gfx1250) — compile-verified, matrix operand order checked by reading, not run on hardware
//
#include <hip/hip_runtime.h>
#include <cstdint>

// ---------------------------------------------------------------------------
// RNNWithMasking for MI455X (gfx1250, wave32, WMMA).
//
// Pipeline:
//   K1 init_ws:        zero h0 + global-barrier words (replay-deterministic)
//   K2 embed_proj:     xw[t][b][g] = emb[x[b,t]] @ W_ih^T + b_ih + b_hh
//                      (bf16 WMMA, LDS-tiled, 128x64 tile per WG)
//   K3 rnn_recurrence: persistent 16-WG cooperative kernel; per step:
//                      h = tanh(xw_t + h @ W_hh^T); masked pool in regs.
//                      W_hh slice pinned in LDS (bf16); h staged via
//                      global_load_async_to_lds_b128 (ASYNCcnt path).
//   K4 fc_out:         logits = rep @ W_fc^T + b_fc (bf16 WMMA, fp32 acc)
//
// Workspace requirement: ~269 MB (xw fp32 dominates).
// ---------------------------------------------------------------------------

#define GLOBAL_AS __attribute__((address_space(1)))
#define LDS_AS    __attribute__((address_space(3)))

typedef __attribute__((ext_vector_type(16))) __bf16 v16bf;
typedef __attribute__((ext_vector_type(8)))  __bf16 v8bf;
typedef __attribute__((ext_vector_type(8)))  float  v8f;
typedef int v4i_b128 __attribute__((vector_size(16)));   // matches builtin param

#if defined(__has_builtin)
#if __has_builtin(__builtin_amdgcn_global_load_async_to_lds_b128) && \
    __has_builtin(__builtin_amdgcn_s_wait_asynccnt)
#define HAVE_ASYNC_LDS 1
#endif
#endif

static constexpr int BATCH = 64;
static constexpr int TSEQ  = 2048;
static constexpr int HID   = 512;
static constexpr int NOUT  = 1000;

// ---- WMMA helpers ----------------------------------------------------------

__device__ __forceinline__ v8f wmma_bf16(v16bf a, v16bf b, v8f c) {
  // D = A(16x32 bf16) x B(32x16 bf16) + C(16x16 f32)
  return __builtin_amdgcn_wmma_f32_16x16x32_bf16(
      /*neg_a=*/false, a, /*neg_b=*/false, b,
      /*c_mod=*/(short)0, c, /*reuse_a=*/false, /*reuse_b=*/false);
}

// Load a 16x32 bf16 fragment (A: rows=M, or B: rows=N since B is K-major per
// row of W) from a row-major [row][stride] bf16 buffer in LDS.
// Lane l: half = l>>4, idx = l&15.  Element j: k = half*8 + j (j<8),
//                                   k = 16 + half*8 + (j-8) (j>=8).
__device__ __forceinline__ v16bf frag_lds(const __bf16* rowbase, int stride, int lane) {
  const int hlf = (lane >> 4) & 1;
  const int idx = lane & 15;
  const __bf16* p = rowbase + idx * stride + hlf * 8;
  v8bf lo = *(const v8bf*)(p);
  v8bf hi = *(const v8bf*)(p + 16);
  v16bf f;
#pragma unroll
  for (int j = 0; j < 8; ++j) { f[j] = lo[j]; f[8 + j] = hi[j]; }
  return f;
}

// ---- K1: init --------------------------------------------------------------

__global__ void init_ws(__bf16* __restrict__ h0, unsigned* __restrict__ bar) {
  int i = blockIdx.x * blockDim.x + threadIdx.x;
  if (i < BATCH * HID) h0[i] = (__bf16)0.0f;
  if (i < 2) bar[i] = 0u;
}

// ---- K2: embedding gather + input projection GEMM --------------------------
// Grid: (BT/128, HID/64). WG = 256 thr (8 waves). Wave w owns M-rows
// [w*16, w*16+16) of the 128-row tile and all 4 N-subtiles (64 cols).

__global__ __launch_bounds__(256, 2) void embed_proj(
    const int*   __restrict__ x,     // [B][T] flattened (bt = b*T + t)
    const float* __restrict__ emb,   // [VOCAB][HID]
    const float* __restrict__ W_ih,  // [HID][HID]
    const float* __restrict__ b_ih,
    const float* __restrict__ b_hh,
    float*       __restrict__ xw)    // [T][B][HID]  (time-major!)
{
  __shared__ __bf16 ldsA[128 * 40];  // 32 cols + 8 pad
  __shared__ __bf16 ldsB[64 * 40];

  const int tid  = threadIdx.x;
  const int lane = tid & 31;
  const int wave = tid >> 5;
  const int m0 = blockIdx.x * 128;   // BT row tile
  const int n0 = blockIdx.y * 64;    // output-feature tile

  // staging roles (fixed per thread)
  const int ar = tid >> 1, ah = tid & 1;   // A: row 0..127, half of 32 cols
  const int bn = tid >> 2, bq = tid & 3;   // B: row 0..63, quarter of 32 cols
  const int tok = x[m0 + ar];              // token for my A row (hoisted)

  v8f acc[4];
#pragma unroll
  for (int i = 0; i < 4; ++i) { v8f z = {}; acc[i] = z; }

  for (int k0 = 0; k0 < HID; k0 += 32) {
    // stage A (embedding gather + f32->bf16)
    {
      const float* src = emb + (size_t)tok * HID + k0 + ah * 16;
      __bf16* dst = ldsA + ar * 40 + ah * 16;
#pragma unroll
      for (int j = 0; j < 16; j += 4) {
        float4 v = *(const float4*)(src + j);
        dst[j + 0] = (__bf16)v.x; dst[j + 1] = (__bf16)v.y;
        dst[j + 2] = (__bf16)v.z; dst[j + 3] = (__bf16)v.w;
      }
    }
    // stage B (W_ih rows n0..n0+63, K-major)
    {
      const float* src = W_ih + (size_t)(n0 + bn) * HID + k0 + bq * 8;
      __bf16* dst = ldsB + bn * 40 + bq * 8;
#pragma unroll
      for (int j = 0; j < 8; j += 4) {
        float4 v = *(const float4*)(src + j);
        dst[j + 0] = (__bf16)v.x; dst[j + 1] = (__bf16)v.y;
        dst[j + 2] = (__bf16)v.z; dst[j + 3] = (__bf16)v.w;
      }
    }
    __syncthreads();
    v16bf a = frag_lds(ldsA + (wave * 16) * 40, 40, lane);
#pragma unroll
    for (int nt = 0; nt < 4; ++nt) {
      v16bf b = frag_lds(ldsB + (nt * 16) * 40, 40, lane);
      acc[nt] = wmma_bf16(a, b, acc[nt]);
    }
    __syncthreads();
  }

  // epilogue: + (b_ih + b_hh), scatter to time-major xw
  const int hlf = (lane >> 4) & 1;
  const int idx = lane & 15;
#pragma unroll
  for (int nt = 0; nt < 4; ++nt) {
    const int g = n0 + nt * 16 + idx;
    const float bias = b_ih[g] + b_hh[g];
#pragma unroll
    for (int r = 0; r < 8; ++r) {
      const int bt = m0 + wave * 16 + r + 8 * hlf;
      const int b  = bt >> 11;          // T = 2048
      const int t  = bt & 2047;
      xw[((size_t)t * BATCH + b) * HID + g] = acc[nt][r] + bias;
    }
  }
}

// ---- K3: persistent cooperative recurrence ---------------------------------

__device__ __forceinline__ void grid_barrier(unsigned* cnt, unsigned* gen,
                                             unsigned nwg) {
  __syncthreads();
  if (threadIdx.x == 0) {
    unsigned g = __atomic_load_n(gen, __ATOMIC_RELAXED);
    if (atomicAdd(cnt, 1u) == nwg - 1) {
      __atomic_store_n(cnt, 0u, __ATOMIC_RELAXED);
      __atomic_fetch_add(gen, 1u, __ATOMIC_RELEASE);
    } else {
      while (__atomic_load_n(gen, __ATOMIC_ACQUIRE) == g)
        __builtin_amdgcn_s_sleep(1);
    }
  }
  __syncthreads();
}

__global__ __launch_bounds__(256, 1) void rnn_recurrence(
    const float* __restrict__ xw,      // [T][B][HID]
    const int*   __restrict__ lengths, // [B]
    const float* __restrict__ W_hh,    // [HID][HID]
    __bf16*      __restrict__ h0,      // [B][HID] ping
    __bf16*      __restrict__ h1,      // [B][HID] pong
    float*       __restrict__ rep,     // [B][HID]
    unsigned*    __restrict__ bar_cnt,
    unsigned*    __restrict__ bar_gen)
{
  __shared__ __bf16 ldsW[32 * 520];  // my 32 rows of W_hh, bf16, K-major
  __shared__ __bf16 ldsH[64 * 520];  // full h, bf16

  const int tid   = threadIdx.x;
  const int lane  = tid & 31;
  const int wave  = tid >> 5;
  const int wg    = blockIdx.x;          // 0..15
  const unsigned nwg = gridDim.x;
  const int g0    = wg * 32;             // my output-column slice
  const int mtile = (wave & 3) * 16;     // 4 M tiles
  const int ntile = (wave >> 2) * 16;    // 2 N tiles within slice

  // Pin my W_hh slice in LDS (bf16), once.
  {
    const int n = tid >> 3, p = tid & 7;          // n:0..31, p:0..7 (64 cols)
    const float* src = W_hh + (size_t)(g0 + n) * HID + p * 64;
    __bf16* dst = ldsW + n * 520 + p * 64;
#pragma unroll
    for (int j = 0; j < 64; j += 4) {
      float4 v = *(const float4*)(src + j);
      dst[j + 0] = (__bf16)v.x; dst[j + 1] = (__bf16)v.y;
      dst[j + 2] = (__bf16)v.z; dst[j + 3] = (__bf16)v.w;
    }
  }

  const int hlf = (lane >> 4) & 1;
  const int idx = lane & 15;
  const int ncol = g0 + ntile + idx;
  int mrow[8]; int lenv[8];
#pragma unroll
  for (int r = 0; r < 8; ++r) {
    mrow[r] = mtile + r + 8 * hlf;       // batch index (M == B)
    lenv[r] = lengths[mrow[r]];
  }

  // h-staging role (fixed per thread): 128 contiguous bf16 = 16 x b128
  const int sm = tid >> 2, sq = tid & 3;

  v8f pool = {};
  __syncthreads();

  for (int t = 0; t < TSEQ; ++t) {
    const __bf16* hprev = (t & 1) ? h1 : h0;
    __bf16*       hnext = (t & 1) ? h0 : h1;

    // stage hprev -> ldsH (async CDNA5 path: global -> LDS without VGPRs)
    {
      const __bf16* src = hprev + sm * HID + sq * 128;
      __bf16* dst = ldsH + sm * 520 + sq * 128;
#ifdef HAVE_ASYNC_LDS
#pragma unroll
      for (int j = 0; j < 128; j += 8)
        __builtin_amdgcn_global_load_async_to_lds_b128(
            (GLOBAL_AS v4i_b128*)(src + j), (LDS_AS v4i_b128*)(dst + j), 0, 0);
      __builtin_amdgcn_s_wait_asynccnt(0);
#else
#pragma unroll
      for (int j = 0; j < 128; j += 8)
        *(v8bf*)(dst + j) = *(const v8bf*)(src + j);
#endif
    }
    __syncthreads();

    v8f acc = {};
#pragma unroll
    for (int k0 = 0; k0 < HID; k0 += 32) {
      v16bf a = frag_lds(ldsH + mtile * 520 + k0, 520, lane);
      v16bf b = frag_lds(ldsW + ntile * 520 + k0, 520, lane);
      acc = wmma_bf16(a, b, acc);
    }

    // epilogue: + xw_t, tanh, masked pool, publish h_next (bf16)
    const float* xrow = xw + (size_t)t * BATCH * HID;
#pragma unroll
    for (int r = 0; r < 8; ++r) {
      float v = acc[r] + xrow[mrow[r] * HID + ncol];
      float h = tanhf(v);
      if (t < lenv[r]) pool[r] += h;
      hnext[mrow[r] * HID + ncol] = (__bf16)h;
    }
    __threadfence();
    grid_barrier(bar_cnt, bar_gen, nwg);
  }

#pragma unroll
  for (int r = 0; r < 8; ++r)
    rep[mrow[r] * HID + ncol] = pool[r];
}

// ---- K4: final FC ----------------------------------------------------------
// Grid: 63 WGs x 128 thr (4 waves). WG owns N tile n0=blockIdx.x*16,
// wave owns M tile m0=wave*16. Converts fp32->bf16 on the fly.

__global__ __launch_bounds__(128, 4) void fc_out(
    const float* __restrict__ rep,    // [B][HID]
    const float* __restrict__ W_fc,   // [NOUT][HID]
    const float* __restrict__ b_fc,   // [NOUT]
    float*       __restrict__ out)    // [B][NOUT]
{
  const int tid = threadIdx.x, lane = tid & 31, wave = tid >> 5;
  const int n0 = blockIdx.x * 16;
  const int m0 = wave * 16;
  const int hlf = (lane >> 4) & 1;
  const int idx = lane & 15;
  const int nrow = n0 + idx;
  const bool bvalid = (nrow < NOUT);
  const float* brow = W_fc + (size_t)(bvalid ? nrow : 0) * HID;

  v8f acc = {};
  for (int k0 = 0; k0 < HID; k0 += 32) {
    const float* ap = rep + (m0 + idx) * HID + k0 + hlf * 8;
    const float* bp = brow + k0 + hlf * 8;
    v16bf a, b;
#pragma unroll
    for (int j = 0; j < 8; ++j) {
      a[j]     = (__bf16)ap[j];
      a[8 + j] = (__bf16)ap[16 + j];
      b[j]     = bvalid ? (__bf16)bp[j]      : (__bf16)0.0f;
      b[8 + j] = bvalid ? (__bf16)bp[16 + j] : (__bf16)0.0f;
    }
    acc = wmma_bf16(a, b, acc);
  }

  const int n = n0 + idx;
  if (n < NOUT) {
    const float bias = b_fc[n];
#pragma unroll
    for (int r = 0; r < 8; ++r) {
      const int m = m0 + r + 8 * hlf;
      out[m * NOUT + n] = acc[r] + bias;
    }
  }
}

// ---- launch ----------------------------------------------------------------

extern "C" void kernel_launch(void* const* d_in, const int* in_sizes, int n_in,
                              void* d_out, int out_size, void* d_ws, size_t ws_size,
                              hipStream_t stream) {
  const int*   x    = (const int*)d_in[0];
  const int*   len  = (const int*)d_in[1];
  const float* emb  = (const float*)d_in[2];
  const float* W_ih = (const float*)d_in[3];
  const float* W_hh = (const float*)d_in[4];
  const float* b_ih = (const float*)d_in[5];
  const float* b_hh = (const float*)d_in[6];
  const float* W_fc = (const float*)d_in[7];
  const float* b_fc = (const float*)d_in[8];
  float* out = (float*)d_out;

  // workspace layout
  char* ws = (char*)d_ws;
  float* xw = (float*)ws;                               // [T][B][HID] fp32
  size_t off = (size_t)TSEQ * BATCH * HID * sizeof(float);
  __bf16* h0 = (__bf16*)(ws + off); off += (size_t)BATCH * HID * 2;
  __bf16* h1 = (__bf16*)(ws + off); off += (size_t)BATCH * HID * 2;
  float* rep = (float*)(ws + off);  off += (size_t)BATCH * HID * sizeof(float);
  off = (off + 255) & ~(size_t)255;
  unsigned* bar = (unsigned*)(ws + off);                // [2]: count, generation

  hipLaunchKernelGGL(init_ws, dim3(128), dim3(256), 0, stream, h0, bar);
  hipLaunchKernelGGL(embed_proj, dim3((BATCH * TSEQ) / 128, HID / 64),
                     dim3(256), 0, stream, x, emb, W_ih, b_ih, b_hh, xw);
  hipLaunchKernelGGL(rnn_recurrence, dim3(16), dim3(256), 0, stream,
                     xw, len, W_hh, h0, h1, rep, bar, bar + 1);
  hipLaunchKernelGGL(fc_out, dim3((NOUT + 15) / 16), dim3(128), 0, stream,
                     rep, W_fc, b_fc, out);
}
